// TemporalDownsampler_17428977287967
// MI455X (gfx1250) — compile-verified, matrix operand order checked by reading, not run on hardware
//
#include <hip/hip_runtime.h>
#include <hip/hip_bf16.h>

// ---------------------------------------------------------------------------
// TemporalDownsampler for MI455X (gfx1250, wave32, WMMA).
//
// Math (derived from the reference):
//   per chunk (b,j), with x0..x2 = x[b, 1+4j .. 3+4j, :]  (x3 unused):
//     y0 = W1t2 . x0 + b1
//     y1 = W1t0 . x0 + W1t1 . x1 + W1t2 . x2 + b1
//     s  = silu(GN1(y0,y1))            (stats over 48ch x 2pos per group)
//     z  = W2t2 . s0 + b2              (only s0, only tap 2)
//     out[b,1+j,:] = silu(GN2(z))      (stats over 48ch per group)
//   out[b,0,:] = x[b,0,:]
//
// Roofline: ~15.1G MACs over ~120MB of HBM traffic -> route GEMMs through
// v_wmma_f32_16x16x32_bf16 (fp32 accumulate). Wave tile 32x64 so each B
// fragment feeds two WMMAs (8 wmma : 12 b128-loads per K-step).
// ---------------------------------------------------------------------------

typedef __attribute__((ext_vector_type(8)))  __bf16 v8bf;
typedef __attribute__((ext_vector_type(16))) __bf16 v16bf;
typedef __attribute__((ext_vector_type(8)))  float  v8f;

#define DD      1536
#define BB      64
#define TT      81
#define NCHUNK  20
#define NS      (BB * NCHUNK)   // 1280 samples
#define K3      (3 * DD)        // 4608

__device__ __forceinline__ unsigned short f2bf(float f) {
  unsigned int u = __float_as_uint(f);
  u += 0x7FFFu + ((u >> 16) & 1u);   // round-to-nearest-even
  return (unsigned short)(u >> 16);
}
__device__ __forceinline__ float silu_f(float x) {
  return x / (1.0f + __expf(-x));
}

// Fragment load per CDNA5 16-bit A/B layout: lane = idx&15 (+16*half),
// half 0 holds k = {0..7, 16..23}, half 1 holds k = {8..15, 24..31};
// two contiguous 16B (b128) loads per lane.
__device__ __forceinline__ v16bf ldfrag(const unsigned short* p) {
  v8bf lo = *(const v8bf*)p;
  v8bf hi = *(const v8bf*)(p + 16);
  return __builtin_shufflevector(lo, hi,
      0,1,2,3,4,5,6,7,8,9,10,11,12,13,14,15);
}

// ---- conversion / staging kernels -----------------------------------------

// Xb[n, t*D + i] = bf16(x[b, 1 + 4j + t, i]),  n = b*20 + j, t = 0..2
__global__ void cvt_x(const float* __restrict__ x, unsigned short* __restrict__ Xb) {
  int idx = blockIdx.x * 256 + threadIdx.x;
  if (idx >= NS * K3) return;
  int n = idx / K3, r = idx % K3;
  int t = r / DD, i = r % DD;
  int b = n / NCHUNK, j = n % NCHUNK;
  Xb[idx] = f2bf(x[(size_t)(b * TT + 1 + 4 * j + t) * DD + i]);
}

// W1c[o, t*D + i] = bf16(w1[o, i, t])  (w1 is (O, I, 3) row-major)
__global__ void cvt_w1(const float* __restrict__ w1, unsigned short* __restrict__ W1c) {
  int idx = blockIdx.x * 256 + threadIdx.x;
  if (idx >= DD * K3) return;
  int o = idx / K3, r = idx % K3;
  int t = r / DD, i = r % DD;
  W1c[idx] = f2bf(w1[((size_t)o * DD + i) * 3 + t]);
}

// W2t[o, i] = bf16(w2[o, i, 2])
__global__ void cvt_w2(const float* __restrict__ w2, unsigned short* __restrict__ W2t) {
  int idx = blockIdx.x * 256 + threadIdx.x;
  if (idx >= DD * DD) return;
  int o = idx / DD, i = idx % DD;
  W2t[idx] = f2bf(w2[((size_t)o * DD + i) * 3 + 2]);
}

// out[b, 0, :] = x[b, 0, :]
__global__ void copy_first(const float* __restrict__ x, float* __restrict__ out) {
  int idx = blockIdx.x * 256 + threadIdx.x;
  if (idx >= BB * DD) return;
  int b = idx / DD, d = idx % DD;
  out[(size_t)b * TT * DD + d] = x[(size_t)b * TT * DD + d];
}

// ---- bf16 WMMA GEMM:  Out[m, n] = sum_k A[m,k] * B[n,k]  (+ bias[n]) -------
// Wave tile: 32(M) x 64(N); block = 8 waves covering 32 x 512 of N; K step 32.
__global__ __launch_bounds__(256) void wmma_gemm(
    const unsigned short* __restrict__ A, int lda,
    const unsigned short* __restrict__ Bw, int ldb,
    const float* __restrict__ bias,
    float* __restrict__ Out, int ldo, int K)
{
  const int lane = threadIdx.x & 31;
  const int wave = threadIdx.x >> 5;
  const int m    = lane & 15;
  const int half = lane >> 4;
  const int rowBase = blockIdx.y * 32;
  const int colBase = (blockIdx.x * 8 + wave) * 64;

  const unsigned short* aP0 = A + (size_t)(rowBase + m) * lda + 8 * half;
  const unsigned short* aP1 = aP0 + (size_t)16 * lda;
  const unsigned short* bP[4];
  #pragma unroll
  for (int t = 0; t < 4; ++t)
    bP[t] = Bw + (size_t)(colBase + t * 16 + m) * ldb + 8 * half;

  v8f zero = {};
  v8f acc[8];                       // [t] = rows 0..15, [4+t] = rows 16..31
  #pragma unroll
  for (int i = 0; i < 8; ++i) acc[i] = zero;

  for (int kk = 0; kk < K; kk += 32) {
    v16bf a0 = ldfrag(aP0 + kk);
    v16bf a1 = ldfrag(aP1 + kk);
    #pragma unroll
    for (int t = 0; t < 4; ++t) {
      v16bf bf = ldfrag(bP[t] + kk);
      acc[t]     = __builtin_amdgcn_wmma_f32_16x16x32_bf16(
          false, a0, false, bf, (short)0, acc[t],     false, false);
      acc[4 + t] = __builtin_amdgcn_wmma_f32_16x16x32_bf16(
          false, a1, false, bf, (short)0, acc[4 + t], false, false);
    }
  }

  // C/D layout: lane (n = lane&15), VGPR v -> row M = v + 8*half.
  #pragma unroll
  for (int t = 0; t < 4; ++t) {
    int col = colBase + t * 16 + m;
    float bv = bias ? bias[col] : 0.0f;
    #pragma unroll
    for (int v = 0; v < 8; ++v) {
      int row = rowBase + v + 8 * half;
      Out[(size_t)row * ldo + col]        = acc[t][v]     + bv;
      Out[(size_t)(row + 16) * ldo + col] = acc[4 + t][v] + bv;
    }
  }
}

// ---- GroupNorm(48ch x 2pos) + SiLU -> S0 (bf16), one block per sample -----
__global__ __launch_bounds__(256) void gn1_silu(
    const float* __restrict__ Y,        // (NS, 2, D): [n*2D + p*D + c]
    const float* __restrict__ gw, const float* __restrict__ gb,
    unsigned short* __restrict__ S0)    // (NS, D)
{
  const int n    = blockIdx.x;
  const int lane = threadIdx.x & 31;
  const int wave = threadIdx.x >> 5;
  const size_t base = (size_t)n * 2 * DD;

  for (int gi = 0; gi < 4; ++gi) {
    const int c0 = (wave * 4 + gi) * 48;
    float s = 0.f, q = 0.f;
    #pragma unroll
    for (int v = lane; v < 96; v += 32) {      // 96 values: 48ch x 2pos
      int p = v / 48, c = c0 + (v % 48);
      float y = Y[base + (size_t)p * DD + c];
      s += y; q += y * y;
    }
    #pragma unroll
    for (int off = 16; off > 0; off >>= 1) {
      s += __shfl_xor(s, off, 32);
      q += __shfl_xor(q, off, 32);
    }
    float mean = s * (1.0f / 96.0f);
    float var  = q * (1.0f / 96.0f) - mean * mean;
    float rs   = rsqrtf(var + 1e-5f);
    // only s0 (p=0) feeds the next conv
    #pragma unroll
    for (int v = lane; v < 48; v += 32) {
      int c = c0 + v;
      float y  = Y[base + c];
      float xn = (y - mean) * rs * gw[c] + gb[c];
      S0[(size_t)n * DD + c] = f2bf(silu_f(xn));
    }
  }
}

// ---- GroupNorm(48ch) + SiLU + scatter to out[b, 1+j, :] -------------------
__global__ __launch_bounds__(256) void gn2_silu(
    const float* __restrict__ Z,        // (NS, D)
    const float* __restrict__ gw, const float* __restrict__ gb,
    float* __restrict__ out)
{
  const int n    = blockIdx.x;
  const int b    = n / NCHUNK, j = n % NCHUNK;
  const int lane = threadIdx.x & 31;
  const int wave = threadIdx.x >> 5;
  const size_t base  = (size_t)n * DD;
  const size_t obase = ((size_t)b * TT + 1 + j) * DD;

  for (int gi = 0; gi < 4; ++gi) {
    const int c0 = (wave * 4 + gi) * 48;
    float s = 0.f, q = 0.f;
    #pragma unroll
    for (int v = lane; v < 48; v += 32) {
      float z = Z[base + c0 + v];
      s += z; q += z * z;
    }
    #pragma unroll
    for (int off = 16; off > 0; off >>= 1) {
      s += __shfl_xor(s, off, 32);
      q += __shfl_xor(q, off, 32);
    }
    float mean = s * (1.0f / 48.0f);
    float var  = q * (1.0f / 48.0f) - mean * mean;
    float rs   = rsqrtf(var + 1e-5f);
    #pragma unroll
    for (int v = lane; v < 48; v += 32) {
      int c = c0 + v;
      float z  = Z[base + c];
      float xn = (z - mean) * rs * gw[c] + gb[c];
      out[obase + c] = silu_f(xn);
    }
  }
}

// ---------------------------------------------------------------------------

extern "C" void kernel_launch(void* const* d_in, const int* in_sizes, int n_in,
                              void* d_out, int out_size, void* d_ws, size_t ws_size,
                              hipStream_t stream) {
  (void)in_sizes; (void)n_in; (void)out_size; (void)ws_size;
  const float* x    = (const float*)d_in[0];
  const float* w1   = (const float*)d_in[1];
  const float* b1   = (const float*)d_in[2];
  const float* gn1w = (const float*)d_in[3];
  const float* gn1b = (const float*)d_in[4];
  const float* w2   = (const float*)d_in[5];
  const float* b2   = (const float*)d_in[6];
  const float* gn2w = (const float*)d_in[7];
  const float* gn2b = (const float*)d_in[8];
  float* out = (float*)d_out;

  // workspace layout (bytes)
  char* ws = (char*)d_ws;
  unsigned short* Xb  = (unsigned short*)(ws);             // 1280*4608*2 = 11,796,480
  unsigned short* W1c = (unsigned short*)(ws + 11796480);  // 1536*4608*2 = 14,155,776
  unsigned short* W2t = (unsigned short*)(ws + 25952256);  // 1536*1536*2 =  4,718,592
  float*          Y   = (float*)         (ws + 30670848);  // 1280*3072*4 = 15,728,640
  unsigned short* S0  = (unsigned short*)(ws + 46399488);  // 1280*1536*2 =  3,932,160
  float*          Z   = (float*)         (ws + 50331648);  // 1280*1536*4 =  7,864,320
                                                           // total ~58.2 MB

  cvt_x <<<(NS * K3 + 255) / 256, 256, 0, stream>>>(x, Xb);
  cvt_w1<<<(DD * K3 + 255) / 256, 256, 0, stream>>>(w1, W1c);
  cvt_w2<<<(DD * DD + 255) / 256, 256, 0, stream>>>(w2, W2t);
  copy_first<<<(BB * DD + 255) / 256, 256, 0, stream>>>(x, out);

  dim3 g(3, 40);  // N: 3 * 512 = 1536, M: 40 * 32 = 1280
  // y1 = [x0|x1|x2] . [W1t0|W1t1|W1t2]^T + b1   (K = 4608)
  wmma_gemm<<<g, 256, 0, stream>>>(Xb, K3, W1c, K3, b1, Y + DD, 2 * DD, K3);
  // y0 = x0 . W1t2^T + b1                       (K = 1536, tap-2 columns)
  wmma_gemm<<<g, 256, 0, stream>>>(Xb, K3, W1c + 2 * DD, K3, b1, Y, 2 * DD, DD);

  gn1_silu<<<NS, 256, 0, stream>>>(Y, gn1w, gn1b, S0);

  // z = s0 . W2t2^T + b2                        (K = 1536)
  wmma_gemm<<<g, 256, 0, stream>>>(S0, DD, W2t, DD, b2, Z, DD, DD);

  gn2_silu<<<NS, 256, 0, stream>>>(Z, gn2w, gn2b, out);
}
